// Head_36455682408807
// MI455X (gfx1250) — compile-verified, matrix operand order checked by reading, not run on hardware
//
#include <hip/hip_runtime.h>
#include <math.h>

// Problem constants (match reference)
#define B_  8
#define T_  2048
#define E_  1024
#define HS_ 64

#define LOG2E 1.44269504088896340736f

typedef __attribute__((ext_vector_type(16))) _Float16 v16h;
typedef __attribute__((ext_vector_type(8)))  _Float16 v8h;
typedef __attribute__((ext_vector_type(8)))  float    v8f;
typedef __attribute__((ext_vector_type(4)))  float    v4f;

// D = A*B + C, f16 inputs, f32 accumulate (v_wmma_f32_16x16x32_f16)
#define WMMA_F16(A, Bf, C) \
  __builtin_amdgcn_wmma_f32_16x16x32_f16(false, (A), false, (Bf), (short)0, (C), false, false)

// Load a 16x32 f16 fragment (A: lane->M / B: lane->N; K packed in halves).
// Per ISA table: lanes 0-15 hold K = {0..7, 16..23}, lanes 16-31 hold K = {8..15, 24..31}.
// Rows contiguous (stride ld halves) -> two 16B loads per lane.
static __device__ __forceinline__ v16h load_frag_f16(const _Float16* p, int ld) {
  const int lane = threadIdx.x & 31;
  const int r    = lane & 15;
  const int h8   = (lane >> 4) * 8;
  const _Float16* row = p + (size_t)r * ld;
  v8h lo = *(const v8h*)(row + h8);
  v8h hi = *(const v8h*)(row + 16 + h8);
  return __builtin_shufflevector(lo, hi, 0,1,2,3,4,5,6,7,8,9,10,11,12,13,14,15);
}

// Raw f32 fragment load: issue all four b128 loads with no dependent ALU in
// between (lets the compiler clause them and keep them in flight together).
static __device__ __forceinline__ void load_raw32(const float* p, int ld, v4f r[4]) {
  const int lane = threadIdx.x & 31;
  const float* row = p + (size_t)(lane & 15) * ld + ((lane >> 4) << 3);
  r[0] = *(const v4f*)(row + 0);
  r[1] = *(const v4f*)(row + 4);
  r[2] = *(const v4f*)(row + 16);
  r[3] = *(const v4f*)(row + 20);
}

static __device__ __forceinline__ v16h cvt_frag(const v4f r[4]) {
  v16h o;
#pragma unroll
  for (int j = 0; j < 4; ++j) {
    o[j]      = (_Float16)r[0][j];
    o[4 + j]  = (_Float16)r[1][j];
    o[8 + j]  = (_Float16)r[2][j];
    o[12 + j] = (_Float16)r[3][j];
  }
  return o;
}

// ---------------------------------------------------------------------------
// Pass 1: q/k/v = x @ W^T + b, output f16.
// q pre-scaled by (1/sqrt(HS)) * log2(e) so pass 2 can use exp2 directly.
// V stored transposed per batch: vtw[b][h][t]  (so P@V B-frags load contiguous).
// Grid: (B*T)/16 blocks, 128 threads (4 waves; wave w owns HS cols 16w..16w+15).
// ---------------------------------------------------------------------------
__global__ void __launch_bounds__(128) qkv_proj_kernel(
    const float* __restrict__ x,
    const float* __restrict__ Wq, const float* __restrict__ bq,
    const float* __restrict__ Wk, const float* __restrict__ bk,
    const float* __restrict__ Wv, const float* __restrict__ bv,
    _Float16* __restrict__ qw, _Float16* __restrict__ kw,
    _Float16* __restrict__ vtw)
{
  const int wave  = threadIdx.x >> 5;
  const int lane  = threadIdx.x & 31;
  const int gt0   = blockIdx.x * 16;        // token tile base (T%16==0 -> no batch straddle)
  const int batch = gt0 / T_;
  const int tloc0 = gt0 - batch * T_;
  const int nb    = wave * 16;              // HS column base for this wave

  const float* xrow = x  + (size_t)gt0 * E_;
  const float* wqr  = Wq + (size_t)nb * E_;
  const float* wkr  = Wk + (size_t)nb * E_;
  const float* wvr  = Wv + (size_t)nb * E_;

  v8f accq = {}, acck = {}, accv = {};
  for (int k0 = 0; k0 < E_; k0 += 32) {
    // Phase 1: issue all 16 global_load_b128 together (overlapped latency)
    v4f ra[4], rq[4], rk[4], rv[4];
    load_raw32(xrow + k0, E_, ra);
    load_raw32(wqr + k0, E_, rq);
    load_raw32(wkr + k0, E_, rk);
    load_raw32(wvr + k0, E_, rv);
    // Phase 2: convert, then 3 chained WMMAs sharing the A fragment
    v16h a  = cvt_frag(ra);
    v16h fq = cvt_frag(rq);
    v16h fk = cvt_frag(rk);
    v16h fv = cvt_frag(rv);
    accq = WMMA_F16(a, fq, accq);
    acck = WMMA_F16(a, fk, acck);
    accv = WMMA_F16(a, fv, accv);
  }

  // C/D layout: lane (n = lane&15, half = lane>>4) holds rows m = half*8 + v.
  const int n  = lane & 15;
  const int mh = (lane >> 4) * 8;
  const float biasq = bq[nb + n];
  const float biask = bk[nb + n];
  const float biasv = bv[nb + n];
  const float qscale = 0.125f * LOG2E;      // 1/sqrt(64) * log2(e)

  v8h vv;
#pragma unroll
  for (int v = 0; v < 8; ++v) {
    const size_t tok = (size_t)(gt0 + mh + v);
    qw[tok * HS_ + nb + n] = (_Float16)((accq[v] + biasq) * qscale);
    kw[tok * HS_ + nb + n] = (_Float16)(acck[v] + biask);
    vv[v] = (_Float16)(accv[v] + biasv);
  }
  // transposed V: 8 consecutive t per lane -> one 16B store
  *(v8h*)(vtw + ((size_t)(batch * HS_ + nb + n)) * T_ + tloc0 + mh) = vv;
}

// ---------------------------------------------------------------------------
// Pass 2: causal flash attention with split-K for occupancy.
// One block per 16-query tile (B*T/16 = 1024 blocks, 4096 waves total).
// The 4 waves partition the key range round-robin (wave w: kb = 32w step 128),
// each keeping private (m, l, O); partials merged through LDS at the end.
// Softmax runs in the base-2 domain (log2e folded into q).
// Row sums come from an extra WMMA against an all-ones B fragment instead of
// cross-lane shuffles (moves work from the VALU critical path onto the XDL).
// ---------------------------------------------------------------------------
__global__ void __launch_bounds__(128) attn_kernel(
    const _Float16* __restrict__ qw, const _Float16* __restrict__ kw,
    const _Float16* __restrict__ vtw, float* __restrict__ out)
{
  __shared__ __align__(16) _Float16 pshare[4][16 * 32]; // per-wave P relayout buffer
  __shared__ __align__(16) float ls_o[4][16][HS_];      // per-wave partial O
  __shared__ float ls_m[4][16];                         // per-wave row max (base-2 domain)
  __shared__ float ls_l[4][16];                         // per-wave row sum

  const int wave  = threadIdx.x >> 5;
  const int lane  = threadIdx.x & 31;
  const int tpb   = T_ / 16;                    // query tiles per batch
  const int batch = blockIdx.x / tpb;
  const int q0    = (blockIdx.x % tpb) * 16;    // in-batch query base
  const int n     = lane & 15;
  const int mh    = (lane >> 4) * 8;

  const _Float16* qb = qw + ((size_t)(batch * T_ + q0)) * HS_;
  const v16h aq0 = load_frag_f16(qb, HS_);      // h = 0..31
  const v16h aq1 = load_frag_f16(qb + 32, HS_); // h = 32..63

  // constant all-ones B fragment for the row-sum WMMA
  v16h ones;
#pragma unroll
  for (int j = 0; j < 16; ++j) ones[j] = (_Float16)1.0f;

  v8f o0 = {}, o1 = {}, o2 = {}, o3 = {};
  float rmax[8], rsum[8];
#pragma unroll
  for (int v = 0; v < 8; ++v) { rmax[v] = -INFINITY; rsum[v] = 0.0f; }

  _Float16* pbuf = pshare[wave];
  const _Float16* vtb = vtw + ((size_t)batch * HS_) * T_;
  const _Float16* kwb = kw + ((size_t)batch * T_) * HS_;

  for (int kb = wave * 32; kb <= q0 + 15; kb += 128) {
    const _Float16* kbp = kwb + (size_t)kb * HS_;

    // Prefetch this wave's next K/V chunk into cache (global_prefetch_b8)
    if (kb + 128 <= q0 + 15) {
      __builtin_prefetch(kbp + (size_t)128 * HS_ + (size_t)lane * HS_, 0, 0);
      __builtin_prefetch(vtb + (size_t)(lane * 2) * T_ + kb + 128, 0, 0);
      __builtin_prefetch(vtb + (size_t)(lane * 2 + 1) * T_ + kb + 128, 0, 0);
    }

    // S[16q x 32k] = Q * K^T  (q carries 1/sqrt(HS) * log2e)
    v8f s0 = {}, s1 = {};
    {
      v16h b0 = load_frag_f16(kbp, HS_);
      v16h b1 = load_frag_f16(kbp + 32, HS_);
      s0 = WMMA_F16(aq0, b0, s0);
      s0 = WMMA_F16(aq1, b1, s0);
    }
    {
      v16h b0 = load_frag_f16(kbp + 16 * HS_, HS_);
      v16h b1 = load_frag_f16(kbp + 16 * HS_ + 32, HS_);
      s1 = WMMA_F16(aq0, b0, s1);
      s1 = WMMA_F16(aq1, b1, s1);
    }

    // Hoist V B-frag loads so they overlap the softmax VALU chain
    v16h bv0 = load_frag_f16(vtb + (size_t)0  * T_ + kb, T_);
    v16h bv1 = load_frag_f16(vtb + (size_t)16 * T_ + kb, T_);
    v16h bv2 = load_frag_f16(vtb + (size_t)32 * T_ + kb, T_);
    v16h bv3 = load_frag_f16(vtb + (size_t)48 * T_ + kb, T_);

    // Online softmax (base-2). Row max reduced across the 16-lane group.
    float sc[8];
    if (kb + 31 > q0) {
      // Diagonal blocks: apply causal mask per element (wave-uniform branch)
#pragma unroll
      for (int v = 0; v < 8; ++v) {
        const int qg = q0 + mh + v;
        const float e0 = (kb + n      > qg) ? -1e30f : s0[v];
        const float e1 = (kb + 16 + n > qg) ? -1e30f : s1[v];
        float mx = fmaxf(e0, e1);
#pragma unroll
        for (int off = 1; off < 16; off <<= 1)
          mx = fmaxf(mx, __shfl_xor(mx, off, 32));
        const float nm = fmaxf(rmax[v], mx);
        sc[v] = __builtin_amdgcn_exp2f(rmax[v] - nm);
        rmax[v] = nm;
        pbuf[(mh + v) * 32 + n]      = (_Float16)__builtin_amdgcn_exp2f(e0 - nm);
        pbuf[(mh + v) * 32 + 16 + n] = (_Float16)__builtin_amdgcn_exp2f(e1 - nm);
      }
    } else {
      // Interior blocks: no masking needed
#pragma unroll
      for (int v = 0; v < 8; ++v) {
        float mx = fmaxf(s0[v], s1[v]);
#pragma unroll
        for (int off = 1; off < 16; off <<= 1)
          mx = fmaxf(mx, __shfl_xor(mx, off, 32));
        const float nm = fmaxf(rmax[v], mx);
        sc[v] = __builtin_amdgcn_exp2f(rmax[v] - nm);
        rmax[v] = nm;
        pbuf[(mh + v) * 32 + n]      = (_Float16)__builtin_amdgcn_exp2f(s0[v] - nm);
        pbuf[(mh + v) * 32 + 16 + n] = (_Float16)__builtin_amdgcn_exp2f(s1[v] - nm);
      }
    }

    // Reload P as an A fragment; row sums via WMMA against ones.
    v16h ap = load_frag_f16(pbuf, 32);
    v8f prow = {};
    prow = WMMA_F16(ap, ones, prow);

    // Rescale running state, fold in new row sums, accumulate O += P @ V
#pragma unroll
    for (int v = 0; v < 8; ++v) {
      rsum[v] = rsum[v] * sc[v] + prow[v];
      o0[v] *= sc[v]; o1[v] *= sc[v]; o2[v] *= sc[v]; o3[v] *= sc[v];
    }
    o0 = WMMA_F16(ap, bv0, o0);
    o1 = WMMA_F16(ap, bv1, o1);
    o2 = WMMA_F16(ap, bv2, o2);
    o3 = WMMA_F16(ap, bv3, o3);
  }

  // Publish per-wave partials (waves with zero iterations publish m=-inf, l=0, O=0)
#pragma unroll
  for (int v = 0; v < 8; ++v) {
    ls_o[wave][mh + v][0 * 16 + n] = o0[v];
    ls_o[wave][mh + v][1 * 16 + n] = o1[v];
    ls_o[wave][mh + v][2 * 16 + n] = o2[v];
    ls_o[wave][mh + v][3 * 16 + n] = o3[v];
  }
  if (n == 0) {
#pragma unroll
    for (int v = 0; v < 8; ++v) {
      ls_m[wave][mh + v] = rmax[v];
      ls_l[wave][mh + v] = rsum[v];
    }
  }
  __syncthreads();

  // Merge the 4 split-K partials: 128 threads x 8 contiguous output cols.
  // Wave 0 always processed kb=0, so the global max is finite.
  const int r  = threadIdx.x >> 3;        // row 0..15
  const int c0 = (threadIdx.x & 7) * 8;   // col base 0..56
  const float gm = fmaxf(fmaxf(ls_m[0][r], ls_m[1][r]),
                         fmaxf(ls_m[2][r], ls_m[3][r]));
  float acc[8] = {0, 0, 0, 0, 0, 0, 0, 0};
  float l = 0.0f;
#pragma unroll
  for (int w = 0; w < 4; ++w) {
    const float e = __builtin_amdgcn_exp2f(ls_m[w][r] - gm);
    l += e * ls_l[w][r];
#pragma unroll
    for (int j = 0; j < 8; ++j) acc[j] += e * ls_o[w][r][c0 + j];
  }
  const float inv = 1.0f / l;
  float* orow = out + ((size_t)(batch * T_ + q0 + r)) * HS_ + c0;
#pragma unroll
  for (int j = 0; j < 8; ++j) orow[j] = acc[j] * inv;
}

extern "C" void kernel_launch(void* const* d_in, const int* in_sizes, int n_in,
                              void* d_out, int out_size, void* d_ws, size_t ws_size,
                              hipStream_t stream) {
  const float* x  = (const float*)d_in[0];
  const float* Wq = (const float*)d_in[1];
  const float* bq = (const float*)d_in[2];
  const float* Wk = (const float*)d_in[3];
  const float* bk = (const float*)d_in[4];
  const float* Wv = (const float*)d_in[5];
  const float* bv = (const float*)d_in[6];
  float* out = (float*)d_out;

  // Workspace: q (f16), k (f16), v-transposed (f16) -> 3 * 2 MB = 6 MB
  _Float16* qw  = (_Float16*)d_ws;
  _Float16* kw  = qw + (size_t)B_ * T_ * HS_;
  _Float16* vtw = kw + (size_t)B_ * T_ * HS_;

  qkv_proj_kernel<<<(B_ * T_) / 16, 128, 0, stream>>>(x, Wq, bq, Wk, bk, Wv, bv,
                                                      qw, kw, vtw);
  attn_kernel<<<(B_ * T_) / 16, 128, 0, stream>>>(qw, kw, vtw, out);
}
